// FMoE_39917426049647
// MI455X (gfx1250) — matrix operand order repaired and checked by reading, not compiled, not run on hardware
//
#include <hip/hip_runtime.h>

#define N_TOK 2048
#define D_ 1024
#define DFF_ 4096
#define E_ 8

typedef __attribute__((ext_vector_type(16))) _Float16 v16h;
typedef __attribute__((ext_vector_type(8)))  _Float16 v8h;
typedef __attribute__((ext_vector_type(4)))  _Float16 v4h;
typedef __attribute__((ext_vector_type(2)))  _Float16 v2h;
typedef __attribute__((ext_vector_type(8)))  float    v8f;
typedef __attribute__((ext_vector_type(4)))  float    v4f;

// ---------------------------------------------------------------------------
// Gate: one wave32 per token. logits = x@Wg + bg (E=8), top-2 + softmax,
// scatter (token, weight) into per-expert lists.
// ---------------------------------------------------------------------------
__global__ __launch_bounds__(256) void moe_gate_kernel(
    const float* __restrict__ x, const float* __restrict__ Wg,
    const float* __restrict__ bg, int* __restrict__ counts,
    int* __restrict__ tok_list, float* __restrict__ wt_list) {
  int gtid = blockIdx.x * blockDim.x + threadIdx.x;
  int tok  = gtid >> 5;
  int lane = threadIdx.x & 31;
  if (tok >= N_TOK) return;

  float acc[E_];
#pragma unroll
  for (int e = 0; e < E_; ++e) acc[e] = 0.f;

  const float* xr = x + (size_t)tok * D_;
  for (int d = lane; d < D_; d += 32) {
    float xv = xr[d];
    const float* wr = Wg + (size_t)d * E_;
#pragma unroll
    for (int e = 0; e < E_; ++e) acc[e] += xv * wr[e];
  }
#pragma unroll
  for (int e = 0; e < E_; ++e) {
#pragma unroll
    for (int off = 16; off > 0; off >>= 1)
      acc[e] += __shfl_xor(acc[e], off, 32);
  }

  if (lane == 0) {
    float lg[E_];
#pragma unroll
    for (int e = 0; e < E_; ++e) lg[e] = acc[e] + bg[e];
    int i0 = 0;
#pragma unroll
    for (int e = 1; e < E_; ++e) if (lg[e] > lg[i0]) i0 = e;
    int i1 = (i0 == 0) ? 1 : 0;
#pragma unroll
    for (int e = 0; e < E_; ++e)
      if (e != i0 && lg[e] > lg[i1]) i1 = e;
    float p1 = 1.f / (1.f + __expf(lg[i0] - lg[i1]));
    float p0 = 1.f - p1;

    int p = atomicAdd(&counts[i0], 1);
    tok_list[i0 * N_TOK + p] = tok;
    wt_list [i0 * N_TOK + p] = p0;
    p = atomicAdd(&counts[i1], 1);
    tok_list[i1 * N_TOK + p] = tok;
    wt_list [i1 * N_TOK + p] = p1;
  }
}

// ---------------------------------------------------------------------------
// Transpose + fp32->f16 convert:  out[e][c][r] = (f16) in[e][r][c]
// Per-expert R x C matrices; 32x32 tiles via LDS; packed b32 stores.
// ---------------------------------------------------------------------------
__global__ __launch_bounds__(256) void transpose_cvt_kernel(
    const float* __restrict__ in, _Float16* __restrict__ out, int R, int C) {
  __shared__ float tile[32][33];
  int bpeC = C >> 5;
  int bpe  = (R >> 5) * bpeC;
  int e    = blockIdx.x / bpe;
  int rem  = blockIdx.x % bpe;
  int r0   = (rem / bpeC) << 5;
  int c0   = (rem % bpeC) << 5;
  const float* I = in  + (size_t)e * R * C;
  _Float16*    O = out + (size_t)e * R * C;

  int tid = threadIdx.x;
  int tc = tid & 31, tr = tid >> 5;
#pragma unroll
  for (int p = 0; p < 4; ++p)
    tile[tr + p * 8][tc] = I[(size_t)(r0 + tr + p * 8) * C + c0 + tc];
  __syncthreads();

  int cc = tid >> 4;          // 0..15
  int rp = (tid & 15) << 1;   // even source-row pair
#pragma unroll
  for (int p = 0; p < 2; ++p) {
    int c = cc + p * 16;
    v2h h;
    h[0] = (_Float16)tile[rp][c];
    h[1] = (_Float16)tile[rp + 1][c];
    *(v2h*)&O[(size_t)(c0 + c) * R + r0 + rp] = h;
  }
}

// ---------------------------------------------------------------------------
// Fragment load helpers.
//   A (16x32 f16): lane L holds row M=L%16, halves {kb..kb+15},
//                  kb = k0 + (L/16)*16  -> two contiguous 16B LDS reads.
//   B (32x16 f16): lane L holds col = c0 + L%16, K = k0 + (L/16)*16 + j
//                  -> 32B contiguous from transposed [N][K] f16 weights.
// ---------------------------------------------------------------------------
__device__ __forceinline__ v16h ldA(const _Float16* row, int kb) {
  v8h lo = *(const v8h*)(row + kb);
  v8h hi = *(const v8h*)(row + kb + 8);
  return __builtin_shufflevector(lo, hi, 0, 1, 2, 3, 4, 5, 6, 7,
                                 8, 9, 10, 11, 12, 13, 14, 15);
}

template <bool USE_TW>
__device__ __forceinline__ v16h ldB(const _Float16* __restrict__ Wt, int ldT,
                                    const float* __restrict__ W, int ldO,
                                    int kbase, int col) {
  if (USE_TW) {
    return *(const v16h*)(Wt + (size_t)col * ldT + kbase);   // 32B contiguous
  } else {
    v16h b;
#pragma unroll
    for (int j = 0; j < 16; ++j)
      b[j] = (_Float16)W[(size_t)(kbase + j) * ldO + col];
    return b;
  }
}

// ---------------------------------------------------------------------------
// Fused expert FFN. Block = 256 threads (8 waves), 32 tokens (2 row-tiles) of
// one expert, one DFF half (split). Grid = E * 64 * 2, uniform early-exit.
//   GEMM1: wave w -> H cols [ch + w*16, +16), 2 WMMAs per 32-K step.
//   GEMM2: wave w -> out cols [w*128, +128), B frag reused by 2 row-tiles.
// ---------------------------------------------------------------------------
template <bool USE_TW>
__global__ __launch_bounds__(256) void moe_ffn_kernel(
    const float* __restrict__ x,  const float* __restrict__ W1,
    const float* __restrict__ b1, const float* __restrict__ W2,
    const float* __restrict__ b2,
    const _Float16* __restrict__ W1t, const _Float16* __restrict__ W2t,
    const int* __restrict__ counts, const int* __restrict__ tok_list,
    const float* __restrict__ wt_list, float* __restrict__ out) {
  __shared__ _Float16 Xa[32][D_ + 8];   // pitch 1032 halves (16B aligned)
  __shared__ _Float16 Hl[32][136];      // pitch 136 halves (16B aligned)
  __shared__ int   s_tok[32];
  __shared__ float s_wt[32];

  const int e     = blockIdx.x >> 7;
  const int rem   = blockIdx.x & 127;
  const int t     = rem >> 1;          // 32-row group within expert
  const int split = rem & 1;           // DFF half
  const int rows  = counts[e];
  if (t * 32 >= rows) return;

  const int tid = threadIdx.x;
  if (tid < 32) {
    int rr = t * 32 + tid;
    if (rr < rows) {
      s_tok[tid] = tok_list[e * N_TOK + rr];
      s_wt[tid]  = wt_list[e * N_TOK + rr];
    } else {
      s_tok[tid] = 0;
      s_wt[tid]  = 0.f;
    }
  }
  __syncthreads();

  // Gather X tile (float4 loads, packed b64 LDS stores).
  for (int idx = tid; idx < 32 * (D_ / 4); idx += 256) {
    int r  = idx >> 8;
    int c4 = (idx & 255) << 2;
    v4f v = *(const v4f*)(x + (size_t)s_tok[r] * D_ + c4);
    v4h h;
#pragma unroll
    for (int q = 0; q < 4; ++q) h[q] = (_Float16)v[q];
    *(v4h*)&Xa[r][c4] = h;
  }
  __syncthreads();

  const int wave = tid >> 5;
  const int lane = tid & 31;
  const int m16  = lane & 15;
  const int hi   = lane >> 4;

  const float*    W1e  = W1  + (size_t)e * D_ * DFF_;
  const float*    W2e  = W2  + (size_t)e * DFF_ * D_;
  const _Float16* W1te = W1t + (size_t)e * D_ * DFF_;
  const _Float16* W2te = W2t + (size_t)e * DFF_ * D_;

  v8f acc0[8], acc1[8];
#pragma unroll
  for (int j = 0; j < 8; ++j) { acc0[j] = (v8f){}; acc1[j] = (v8f){}; }

  const int chBase = split * (DFF_ / 2);
  for (int ci = 0; ci < (DFF_ / 2) / 128; ++ci) {
    const int ch   = chBase + ci * 128;
    const int colH = wave * 16 + m16;

    // ---- GEMM1: H[:, ch+colH] over K = D ----
    v8f h0 = (v8f){}, h1 = (v8f){};
#pragma unroll 2
    for (int k0 = 0; k0 < D_; k0 += 32) {
      v16h b  = ldB<USE_TW>(W1te, D_, W1e, DFF_, k0 + hi * 16, ch + colH);
      int kbA = k0 + hi * 16;
      v16h a0 = ldA(&Xa[m16][0], kbA);
      v16h a1 = ldA(&Xa[m16 + 16][0], kbA);
      h0 = __builtin_amdgcn_wmma_f32_16x16x32_f16(false, a0, false, b,
                                                  (short)0, h0, false, false);
      h1 = __builtin_amdgcn_wmma_f32_16x16x32_f16(false, a1, false, b,
                                                  (short)0, h1, false, false);
    }
    {
      float bb = b1[e * DFF_ + ch + colH];
#pragma unroll
      for (int r = 0; r < 8; ++r) {
        float v0 = h0[r] + bb;
        float v1 = h1[r] + bb;
        Hl[hi * 8 + r][colH]      = (_Float16)(v0 > 0.f ? v0 : 0.f);
        Hl[16 + hi * 8 + r][colH] = (_Float16)(v1 > 0.f ? v1 : 0.f);
      }
    }
    __syncthreads();

    // ---- GEMM2: out[:, wave*128 .. +128) += H_chunk @ W2[ch.., :] ----
#pragma unroll
    for (int kk = 0; kk < 128; kk += 32) {
      int kbA = kk + hi * 16;
      v16h a0 = ldA(&Hl[m16][0], kbA);
      v16h a1 = ldA(&Hl[m16 + 16][0], kbA);
      int kbB = ch + kk + hi * 16;
#pragma unroll
      for (int j = 0; j < 8; ++j) {
        int col = wave * 128 + j * 16 + m16;
        v16h b = ldB<USE_TW>(W2te, DFF_, W2e, D_, kbB, col);
        acc0[j] = __builtin_amdgcn_wmma_f32_16x16x32_f16(
            false, a0, false, b, (short)0, acc0[j], false, false);
        acc1[j] = __builtin_amdgcn_wmma_f32_16x16x32_f16(
            false, a1, false, b, (short)0, acc1[j], false, false);
      }
    }
    __syncthreads();   // Hl reused next chunk
  }

  // ---- Epilogue: weighted combine (+ b2 once, on split 0) ----
#pragma unroll
  for (int j = 0; j < 8; ++j) {
    int col = wave * 128 + j * 16 + m16;
    float bb2 = (split == 0) ? b2[e * D_ + col] : 0.f;
#pragma unroll
    for (int r = 0; r < 8; ++r) {
      int M0 = hi * 8 + r;
      int M1 = 16 + M0;
      atomicAdd(&out[(size_t)s_tok[M0] * D_ + col],
                s_wt[M0] * (acc0[j][r] + bb2));
      atomicAdd(&out[(size_t)s_tok[M1] * D_ + col],
                s_wt[M1] * (acc1[j][r] + bb2));
    }
  }
}

// ---------------------------------------------------------------------------
extern "C" void kernel_launch(void* const* d_in, const int* in_sizes, int n_in,
                              void* d_out, int out_size, void* d_ws, size_t ws_size,
                              hipStream_t stream) {
  const float* x  = (const float*)d_in[0];
  const float* Wg = (const float*)d_in[1];
  const float* bg = (const float*)d_in[2];
  const float* W1 = (const float*)d_in[3];
  const float* b1 = (const float*)d_in[4];
  const float* W2 = (const float*)d_in[5];
  const float* b2 = (const float*)d_in[6];
  float* out = (float*)d_out;

  // ws layout: [0,64) counts | [64,+64K) tok_list | [+64K,+128K) wt_list
  //            [256K, +64M) W1t (f16, [E][DFF][D])
  //            [256K+64M, +64M) W2t (f16, [E][D][DFF])
  const size_t OFF_LISTS = 64;
  const size_t OFF_W1T   = 256 * 1024;
  const size_t W_HALVES  = (size_t)E_ * D_ * DFF_;            // 33.5M halves
  const size_t OFF_W2T   = OFF_W1T + W_HALVES * 2;
  const size_t REQUIRED  = OFF_W2T + W_HALVES * 2;

  int*      counts   = (int*)d_ws;
  int*      tok_list = (int*)((char*)d_ws + OFF_LISTS);
  float*    wt_list  = (float*)((char*)d_ws + OFF_LISTS + (size_t)E_ * N_TOK * 4);
  _Float16* W1t      = (_Float16*)((char*)d_ws + OFF_W1T);
  _Float16* W2t      = (_Float16*)((char*)d_ws + OFF_W2T);

  const bool use_tw = (ws_size >= REQUIRED);

  hipMemsetAsync(d_ws, 0, 64, stream);
  hipMemsetAsync(d_out, 0, (size_t)out_size * sizeof(float), stream);

  moe_gate_kernel<<<(N_TOK * 32) / 256, 256, 0, stream>>>(
      x, Wg, bg, counts, tok_list, wt_list);

  if (use_tw) {
    transpose_cvt_kernel<<<E_ * (D_ / 32) * (DFF_ / 32), 256, 0, stream>>>(
        W1, W1t, D_, DFF_);
    transpose_cvt_kernel<<<E_ * (DFF_ / 32) * (D_ / 32), 256, 0, stream>>>(
        W2, W2t, DFF_, D_);
    moe_ffn_kernel<true><<<E_ * 128, 256, 0, stream>>>(
        x, W1, b1, W2, b2, W1t, W2t, counts, tok_list, wt_list, out);
  } else {
    moe_ffn_kernel<false><<<E_ * 128, 256, 0, stream>>>(
        x, W1, b1, W2, b2, W1t, W2t, counts, tok_list, wt_list, out);
  }
}
